// RecurrentNetwork_89172111000316
// MI455X (gfx1250) — matrix operand order repaired and checked by reading, not compiled, not run on hardware
//
#include <hip/hip_runtime.h>

// Problem sizes (fixed by the reference)
#define TT 256
#define BB 256
#define DD 512
#define HH 512
#define OO 512
#define H3 1536   // 3*H

typedef __attribute__((ext_vector_type(16))) __bf16 bf16x16;
typedef __attribute__((ext_vector_type(8)))  float  f32x8;

// fp32 -> bf16 bits, round-to-nearest-even
__device__ __forceinline__ unsigned short f2bf_bits(float f) {
  unsigned u = __builtin_bit_cast(unsigned, f);
  unsigned r = u + 0x7FFFu + ((u >> 16) & 1u);
  return (unsigned short)(r >> 16);
}

// two fp32 -> packed bf16x2 (lo in [15:0], hi in [31:16])
__device__ __forceinline__ unsigned pack2bf(float lo, float hi) {
  return (unsigned)f2bf_bits(lo) | ((unsigned)f2bf_bits(hi) << 16);
}

struct alignas(16) FragBits { uint4 lo, hi; };

// Build a 16x bf16 fragment from two 16-byte chunks in LDS
__device__ __forceinline__ bf16x16 make_frag(const unsigned short* p_lo,
                                             const unsigned short* p_hi) {
  FragBits fb;
  fb.lo = *(const uint4*)p_lo;
  fb.hi = *(const uint4*)p_hi;
  return __builtin_bit_cast(bf16x16, fb);
}

__device__ __forceinline__ void wait_async0() {
#if __has_builtin(__builtin_amdgcn_s_wait_asynccnt)
  __builtin_amdgcn_s_wait_asynccnt(0);
#else
  asm volatile("s_wait_asynccnt 0x0" ::: "memory");
#endif
}

// ---------------------------------------------------------------------------
// Generic GEMM:  C[M,N] = act(A[M,K] @ W[K,N] + bias[N])
//   A: fp32 or bf16 (row-major), W/bias: fp32, C: fp32 or bf16.
//   Block = 256 threads (8 waves). Block tile = 64(M) x 64(N).
//   Each wave owns two 16x16 tiles -> 2 WMMAs per A fragment per K-step.
//   bf16 A tiles are staged with GLOBAL_LOAD_ASYNC_TO_LDS_B128 (ASYNCcnt).
// ---------------------------------------------------------------------------
template<bool A_BF16, bool RELU, bool OUT_BF16>
__global__ __launch_bounds__(256)
void wmma_gemm_kernel(const void* __restrict__ Ain,
                      const float* __restrict__ W,
                      const float* __restrict__ bias,
                      void* __restrict__ Cout,
                      int M, int N, int K) {
  // +8 pad keeps 16B alignment of sub-fragments and spreads LDS banks
  __shared__ alignas(16) unsigned short sA[64][40];   // [m][k]  (bf16 bits)
  __shared__ alignas(16) unsigned short sB[64][40];   // [n][k]  transposed

  const int tid  = threadIdx.x;
  const int lane = tid & 31;
  const int wave = tid >> 5;          // 0..7
  const int mt   = wave >> 1;         // 0..3  (16-row tile)
  const int ng   = wave & 1;          // 0..1  (32-col group)
  const int r0   = blockIdx.y * 64;
  const int n0   = blockIdx.x * 64;

  const int nl  = lane & 15;
  const int cn0 = n0 + ng * 32 + nl;        // tile-0 column
  const int cn1 = cn0 + 16;                 // tile-1 column
  f32x8 acc0 = {}, acc1 = {};
  {
    float bv0 = bias[cn0], bv1 = bias[cn1];
#pragma unroll
    for (int v = 0; v < 8; ++v) { acc0[v] = bv0; acc1[v] = bv1; }
  }

  // staging roles
  const int ar  = tid >> 2;           // 0..63 : A row
  const int ac  = (tid & 3) << 3;     // 0,8,16,24 : A col (8 elems)
  const int bkp = tid >> 4;           // 0..15 : W k-pair (k = 2*bkp, 2*bkp+1)
  const int bn  = (tid & 15) << 2;    // 0..60 : W n-col (4 elems)

  // A fragments: lanes 0-15 hold K 0-15, lanes 16-31 hold K 16-31, M = lane&15
  const unsigned short* pA  = &sA[mt * 16 + nl][(lane < 16) ? 0 : 16];
  // B fragments: lanes 0-15 hold K 0-15, lanes 16-31 hold K 16-31, N = lane&15
  const unsigned short* pB0 = &sB[ng * 32 + nl][(lane < 16) ? 0 : 16];
  const unsigned short* pB1 = pB0 + 16 * 40;

  unsigned* sA32 = (unsigned*)&sA[0][0];    // row stride 20 uints
  unsigned* sB32 = (unsigned*)&sB[0][0];    // row stride 20 uints

  for (int k0 = 0; k0 < K; k0 += 32) {
    // ---- stage A tile [64 x 32] ----
    if constexpr (A_BF16) {
      // async DMA: 16 bytes (8 bf16) per lane, global -> LDS, no VGPR transit
      const unsigned short* Abf = (const unsigned short*)Ain;
      unsigned ldsA = (unsigned)(unsigned long long)&sA[ar][ac];
      unsigned goff = (unsigned)(((size_t)(r0 + ar) * K + k0 + ac) * 2u);
      asm volatile("global_load_async_to_lds_b128 %0, %1, %2"
                   :: "v"(ldsA), "v"(goff), "s"(Abf)
                   : "memory");
    } else {
      const float* Af = (const float*)Ain;
      const float* ap = Af + (size_t)(r0 + ar) * K + k0 + ac;
      float4 f0 = *(const float4*)ap;
      float4 f1 = *(const float4*)(ap + 4);
      const int ab = ar * 20 + (ac >> 1);
      sA32[ab + 0] = pack2bf(f0.x, f0.y);
      sA32[ab + 1] = pack2bf(f0.z, f0.w);
      sA32[ab + 2] = pack2bf(f1.x, f1.y);
      sA32[ab + 3] = pack2bf(f1.z, f1.w);
    }
    // ---- stage W tile [32k x 64n], transposed into sB[n][k], k-pair packed ----
    {
      const float* wp = W + (size_t)(k0 + 2 * bkp) * N + n0 + bn;
      float4 wa = *(const float4*)wp;        // k even
      float4 wb = *(const float4*)(wp + N);  // k odd
      sB32[(bn + 0) * 20 + bkp] = pack2bf(wa.x, wb.x);
      sB32[(bn + 1) * 20 + bkp] = pack2bf(wa.y, wb.y);
      sB32[(bn + 2) * 20 + bkp] = pack2bf(wa.z, wb.z);
      sB32[(bn + 3) * 20 + bkp] = pack2bf(wa.w, wb.w);
    }
    // prefetch next weight tile into cache (global_prefetch_b8)
    if (lane == 0 && k0 + 32 < K) {
      __builtin_prefetch((const void*)(W + (size_t)(k0 + 32) * N + n0), 0, 1);
    }
    if constexpr (A_BF16) wait_async0();
    __syncthreads();

    bf16x16 a  = make_frag(pA, pA + 8);
    bf16x16 b0 = make_frag(pB0, pB0 + 8);
    bf16x16 b1 = make_frag(pB1, pB1 + 8);
    acc0 = __builtin_amdgcn_wmma_f32_16x16x32_bf16(
        false, a, false, b0, (short)0, acc0, false, false);
    acc1 = __builtin_amdgcn_wmma_f32_16x16x32_bf16(
        false, a, false, b1, (short)0, acc1, false, false);
    __syncthreads();
  }

  // C/D layout: VGPR v -> M = v + (lane>=16 ? 8 : 0), N = lane & 15
  const int mrow = r0 + mt * 16 + ((lane >> 4) << 3);
#pragma unroll
  for (int v = 0; v < 8; ++v) {
    float v0 = acc0[v], v1 = acc1[v];
    if (RELU) { v0 = fmaxf(v0, 0.0f); v1 = fmaxf(v1, 0.0f); }
    size_t row = (size_t)(mrow + v) * N;
    if (OUT_BF16) {
      ((unsigned short*)Cout)[row + cn0] = f2bf_bits(v0);
      ((unsigned short*)Cout)[row + cn1] = f2bf_bits(v1);
    } else {
      ((float*)Cout)[row + cn0] = v0;
      ((float*)Cout)[row + cn1] = v1;
    }
  }
}

// ---------------------------------------------------------------------------
// Fused GRU step (one launch per time-step):
//   h_masked = done ? 0 : h_in
//   gh = h_masked @ Wh               (WMMA, 3 gate accumulators per tile)
//   r  = sigmoid(gi_r + gh_r); z = sigmoid(gi_z + gh_z)
//   n  = tanh(gi_n + r*(gh_n + bhn))
//   h_out = (1-z)*n + z*h_masked ;  y_t = bf16(h_out)
// Grid: (H/128, B/16), 256 threads. Each of the 8 waves owns one 16-col tile.
// ---------------------------------------------------------------------------
__global__ __launch_bounds__(256)
void gru_step_kernel(const float* __restrict__ h_in,
                     float* __restrict__ h_out,
                     const unsigned char* __restrict__ done_t,   // [B] bool
                     const float* __restrict__ gi_t,             // [B, 3H]
                     const float* __restrict__ Wh,               // [H, 3H]
                     const float* __restrict__ bhn,              // [H]
                     unsigned short* __restrict__ y_t) {         // [B, H] bf16
  __shared__ alignas(16) unsigned short sA[16][40];         // h tile (bf16)
  __shared__ alignas(16) unsigned short sB[24][16][40];     // [gate*8+wave][n][k]
  __shared__ float srst[16];                                // 0 if done else 1

  const int tid  = threadIdx.x;
  const int lane = tid & 31;
  const int wave = tid >> 5;           // 0..7
  const int b0   = blockIdx.y * 16;
  const int n0   = blockIdx.x * 128;

  if (tid < 16) srst[tid] = done_t[b0 + tid] ? 0.0f : 1.0f;
  __syncthreads();

  f32x8 ar_ = {}, az_ = {}, an_ = {};

  const int sr  = tid >> 4;            // 0..15 : h row
  const int sc  = (tid & 15) << 1;     // 0..30 : h col pair
  const int bnn = tid & 15;            // 0..15 : Wh n-col
  const int bkp = tid >> 4;            // 0..15 : Wh k-pair

  const unsigned short* pA = &sA[lane & 15][(lane < 16) ? 0 : 16];
  const int nl = lane & 15;
  const int kb = (lane < 16) ? 0 : 16;

  unsigned* sA32 = (unsigned*)&sA[0][0];        // row stride 20 uints
  unsigned* sB32 = (unsigned*)&sB[0][0][0];     // [p][n][kp]: (p*16+n)*20+kp

  for (int k0 = 0; k0 < HH; k0 += 32) {
    // stage masked h tile [16 x 32], packed bf16x2 stores
    {
      float m = srst[sr];
      const float* hp = h_in + (size_t)(b0 + sr) * HH + k0 + sc;
      float2 hv = *(const float2*)hp;
      sA32[sr * 20 + (sc >> 1)] = pack2bf(hv.x * m, hv.y * m);
    }
    // stage 24 Wh sub-tiles (3 gates x 8 wave-columns), transposed [n][k],
    // k-pair packed (one ds_store_b32 per thread per sub-tile)
#pragma unroll
    for (int p = 0; p < 24; ++p) {
      const int g  = p >> 3;           // gate 0..2
      const int wv = p & 7;            // wave col tile
      const float* wp = Wh + (size_t)(k0 + 2 * bkp) * H3 + g * HH + n0 + wv * 16 + bnn;
      float w0 = wp[0];
      float w1 = wp[H3];               // next k row
      sB32[(p * 16 + bnn) * 20 + bkp] = pack2bf(w0, w1);
    }
    if (lane == 0 && k0 + 32 < HH) {
      __builtin_prefetch((const void*)(Wh + (size_t)(k0 + 32) * H3 + n0), 0, 1);
    }
    __syncthreads();

    bf16x16 a = make_frag(pA, pA + 8);
    const unsigned short* pr = &sB[0 * 8 + wave][nl][kb];
    const unsigned short* pz = &sB[1 * 8 + wave][nl][kb];
    const unsigned short* pn = &sB[2 * 8 + wave][nl][kb];
    bf16x16 br = make_frag(pr, pr + 8);
    bf16x16 bz = make_frag(pz, pz + 8);
    bf16x16 bn = make_frag(pn, pn + 8);
    ar_ = __builtin_amdgcn_wmma_f32_16x16x32_bf16(false, a, false, br, (short)0, ar_, false, false);
    az_ = __builtin_amdgcn_wmma_f32_16x16x32_bf16(false, a, false, bz, (short)0, az_, false, false);
    an_ = __builtin_amdgcn_wmma_f32_16x16x32_bf16(false, a, false, bn, (short)0, an_, false, false);
    __syncthreads();
  }

  // gates + state update; C layout: M = v + (lane>=16?8:0), N = lane&15
  const int col = n0 + wave * 16 + nl;
  const int mb  = (lane >> 4) << 3;
  const float bh = bhn[col];
#pragma unroll
  for (int v = 0; v < 8; ++v) {
    const int b = b0 + mb + v;
    const float* gir = gi_t + (size_t)b * H3;
    float r  = 1.0f / (1.0f + __expf(-(gir[col]        + ar_[v])));
    float z  = 1.0f / (1.0f + __expf(-(gir[HH + col]   + az_[v])));
    float nn = tanhf(gir[2 * HH + col] + r * (an_[v] + bh));
    float hprev = h_in[(size_t)b * HH + col] * srst[mb + v];
    float nh = (1.0f - z) * nn + z * hprev;
    h_out[(size_t)b * HH + col] = nh;
    y_t[(size_t)b * HH + col]   = f2bf_bits(nh);
  }
}

// ---------------------------------------------------------------------------
// Host launcher
// ---------------------------------------------------------------------------
extern "C" void kernel_launch(void* const* d_in, const int* in_sizes, int n_in,
                              void* d_out, int out_size, void* d_ws, size_t ws_size,
                              hipStream_t stream) {
  (void)in_sizes; (void)n_in; (void)out_size; (void)ws_size;
  const float* hidden = (const float*)d_in[0];
  const float* obs    = (const float*)d_in[1];
  const unsigned char* dones = (const unsigned char*)d_in[2];  // jax bool = 1 byte
  const float* W1  = (const float*)d_in[3];
  const float* b1  = (const float*)d_in[4];
  const float* W2  = (const float*)d_in[5];
  const float* b2  = (const float*)d_in[6];
  const float* W3  = (const float*)d_in[7];
  const float* b3  = (const float*)d_in[8];
  const float* Wi  = (const float*)d_in[9];
  const float* bi  = (const float*)d_in[10];
  const float* Wh  = (const float*)d_in[11];
  const float* bhn = (const float*)d_in[12];
  const float* Wo  = (const float*)d_in[13];
  const float* bo  = (const float*)d_in[14];
  float* out = (float*)d_out;   // [B*H final hidden][T*B*O output]

  const int MTB = TT * BB;      // 65536 rows for all batched GEMMs

  // workspace layout (bytes)
  char* ws = (char*)d_ws;
  unsigned short* x1 = (unsigned short*)(ws);                 //  64 MB bf16
  unsigned short* x2 = (unsigned short*)(ws + 67108864ull);   //  64 MB bf16
  float*          gi = (float*)(ws + 134217728ull);           // 403 MB fp32
  unsigned short* y  = (unsigned short*)(ws + 536870912ull);  //  64 MB bf16
  float*          h0 = (float*)(ws + 603979776ull);           // 512 KB
  float*          h1 = (float*)(ws + 604504064ull);           // 512 KB

  hipMemcpyAsync(h0, hidden, (size_t)BB * HH * sizeof(float),
                 hipMemcpyDeviceToDevice, stream);

  dim3 blk(256);
  // --- MLP encoder (3 layers, relu) ---
  wmma_gemm_kernel<false, true, true><<<dim3(HH / 64, MTB / 64), blk, 0, stream>>>(
      obs, W1, b1, x1, MTB, HH, DD);
  wmma_gemm_kernel<true, true, true><<<dim3(HH / 64, MTB / 64), blk, 0, stream>>>(
      x1, W2, b2, x2, MTB, HH, HH);
  wmma_gemm_kernel<true, true, true><<<dim3(HH / 64, MTB / 64), blk, 0, stream>>>(
      x2, W3, b3, x1, MTB, HH, HH);
  // --- hoisted input-gate GEMM: gi = x3 @ Wi + bi  (independent of carry) ---
  wmma_gemm_kernel<true, false, false><<<dim3(H3 / 64, MTB / 64), blk, 0, stream>>>(
      x1, Wi, bi, gi, MTB, H3, HH);

  // --- sequential GRU scan: one fused launch per step, h ping-pong ---
  float* hp[2] = {h0, h1};
  for (int t = 0; t < TT; ++t) {
    gru_step_kernel<<<dim3(HH / 128, BB / 16), blk, 0, stream>>>(
        hp[t & 1], hp[(t + 1) & 1],
        dones + (size_t)t * BB,
        gi + (size_t)t * BB * H3,
        Wh, bhn,
        y + (size_t)t * BB * HH);
  }
  // after t=255, final hidden lives in hp[(255+1)&1] == h0
  hipMemcpyAsync(out, h0, (size_t)BB * HH * sizeof(float),
                 hipMemcpyDeviceToDevice, stream);

  // --- output GEMM: out = y @ Wo + bo ---
  wmma_gemm_kernel<true, false, false><<<dim3(OO / 64, MTB / 64), blk, 0, stream>>>(
      y, Wo, bo, out + (size_t)BB * HH, MTB, OO, HH);
}
